// FeatureExtractor_88338887344847
// MI455X (gfx1250) — compile-verified, matrix operand order checked by reading, not run on hardware
//
#include <hip/hip_runtime.h>

// ---------------------------------------------------------------------------
// CDNA5 (gfx1250) feature extractor:
//   conv1(12->64,3x3)+ReLU -> maxpool2x2 -> conv2(64->128,3x3)+ReLU
//   + position-embedding -> per-ROI masked spatial max.
// Both convs run as implicit GEMM on v_wmma_f32_16x16x32_f16 with operands
// pre-packed / im2col'd so every WMMA operand is fetched with b128 loads.
// Working set (~25MB) is L2-resident (192MB L2), so scattered im2col gathers
// are cheap; the GEMM inner loops are WMMA-dense and fully unrolled so all
// addresses are base+immediate.
// ---------------------------------------------------------------------------

typedef __attribute__((ext_vector_type(16))) _Float16 v16h;
typedef __attribute__((ext_vector_type(8)))  _Float16 v8h;
typedef __attribute__((ext_vector_type(8)))  float    v8f;
typedef __attribute__((ext_vector_type(4)))  float    v4f;

#define BATCH 16
#define CIN   12          // 3*NF
#define SZ0   128
#define N1    (SZ0 * SZ0) // 16384 conv1 output pixels
#define C1CH  64
#define HALFW 64
#define N2    (HALFW * HALFW) // 4096 pooled pixels
#define C2CH  128
#define K1V   108         // 12*9
#define K1P   128         // padded to 4 WMMA K-steps
#define KT1   4
#define K2V   576         // 64*9  (= 18 * 32, no padding needed)
#define KT2   18
#define NOBJ  32

// ISA 7.12.2: 16-bit A/B operand half-index -> K offset within a 32-K step.
__device__ __forceinline__ int wmma_kk(int lane, int e) {
  return ((lane >> 4) << 3) + (e & 7) + ((e >> 3) << 4);
}

__device__ __forceinline__ v8f wmma16x16x32(v16h a, v16h b, v8f c) {
  // (neg_a, A, neg_b, B, c_mod, C, reuse_a, reuse_b)
  return __builtin_amdgcn_wmma_f32_16x16x32_f16(false, a, false, b, (short)0, c,
                                                false, false);
}

__device__ __forceinline__ v16h cat8(v8h lo, v8h hi) {
  return __builtin_shufflevector(lo, hi, 0, 1, 2, 3, 4, 5, 6, 7, 8, 9, 10, 11,
                                 12, 13, 14, 15);
}

// --------------------------- weight packing --------------------------------
// A-operand layout [mtile][ktile][lane][e] so each lane loads 32 contiguous B.
__global__ void pack_w1(const float* __restrict__ w1, _Float16* __restrict__ a1p) {
  int t = blockIdx.x * blockDim.x + threadIdx.x;       // 4*KT1*32*16 = 8192
  int e = t & 15, lane = (t >> 4) & 31, kt = (t >> 9) & 3, mt = t >> 11;
  int K = kt * 32 + wmma_kk(lane, e);
  int m = mt * 16 + (lane & 15);
  float v = 0.f;
  if (K < K1V) {
    int c = K / 9, r = K % 9, fy = r / 3, fx = r % 3;
    v = w1[((m * CIN + c) * 3 + fy) * 3 + fx];
  }
  a1p[t] = (_Float16)v;
}

__global__ void pack_w2(const float* __restrict__ w2, _Float16* __restrict__ a2p) {
  int t = blockIdx.x * blockDim.x + threadIdx.x;       // 8*KT2*32*16 = 73728
  int e = t & 15, lane = (t >> 4) & 31;
  int g = t >> 9;
  int kt = g % KT2, mt = g / KT2;
  int K = kt * 32 + wmma_kk(lane, e);
  int m = mt * 16 + (lane & 15);
  int c = K / 9, r = K % 9, fy = r / 3, fx = r % 3;
  a2p[t] = (_Float16)w2[((m * C1CH + c) * 3 + fy) * 3 + fx];
}

// ------------------------ position embedding -------------------------------
// PE[p][c] f32, shared across batch. grid = [gy,gx,1-gy,1-gx], step 1/63.
__global__ void pe_kernel(const float* __restrict__ dw, const float* __restrict__ db,
                          float* __restrict__ pe) {
  int t = blockIdx.x * blockDim.x + threadIdx.x;       // N2*C2CH
  int c = t & 127, p = t >> 7;
  float gy = (float)(p >> 6) * (1.0f / 63.0f);
  float gx = (float)(p & 63) * (1.0f / 63.0f);
  float g0 = gy, g1 = gx, g2 = 1.0f - gy, g3 = 1.0f - gx;
  const float* w = dw + c * 4;
  pe[t] = db[c] + g0 * w[0] + g1 * w[1] + g2 * w[2] + g3 * w[3];
}

// ------------------------------ im2col -------------------------------------
// IC1[p][K1P] f16, one thread per 8-K run -> single b128 store.
__global__ void im2col1(const float* __restrict__ img, _Float16* __restrict__ ic) {
  int t = blockIdx.x * blockDim.x + threadIdx.x;       // N1*16
  int run = t & 15, p = t >> 4;
  int y = p >> 7, x = p & 127, K0 = run * 8;
  v8h v;
#pragma unroll
  for (int j = 0; j < 8; ++j) {
    int K = K0 + j;
    float f = 0.f;
    if (K < K1V) {
      int c = K / 9, r = K % 9;
      int iy = y + r / 3 - 1, ix = x + r % 3 - 1;
      if (iy >= 0 && iy < SZ0 && ix >= 0 && ix < SZ0)
        f = img[(c * SZ0 + iy) * SZ0 + ix];
    }
    v[j] = (_Float16)f;
  }
  *(v8h*)(ic + p * K1P + K0) = v;
}

// IC2[p][K2V] f16 from pooled features P[c][y][x] (f16).
__global__ void im2col2(const _Float16* __restrict__ pf, _Float16* __restrict__ ic) {
  int t = blockIdx.x * blockDim.x + threadIdx.x;       // N2*72
  int run = t % 72, p = t / 72;
  int y = p >> 6, x = p & 63, K0 = run * 8;
  v8h v;
#pragma unroll
  for (int j = 0; j < 8; ++j) {
    int K = K0 + j;
    int c = K / 9, r = K % 9;
    int iy = y + r / 3 - 1, ix = x + r % 3 - 1;
    _Float16 f = (_Float16)0.f;
    if (iy >= 0 && iy < HALFW && ix >= 0 && ix < HALFW)
      f = pf[c * N2 + iy * HALFW + ix];
    v[j] = f;
  }
  *(v8h*)(ic + p * K2V + K0) = v;
}

// ------------------------------ GEMM 1 -------------------------------------
// D[64ch x 16384px] = A1[64x128] * IC1[128x16384]; 2M x 2N tile per wave.
// Epilogue: +bias, ReLU, f16 store to C1[p][64c].
__global__ void gemm1(const _Float16* __restrict__ a1p, const _Float16* __restrict__ ic,
                      const float* __restrict__ b1, _Float16* __restrict__ c1) {
  int wave = (blockIdx.x * blockDim.x + threadIdx.x) >> 5;  // 1024 waves
  int lane = threadIdx.x & 31;
  int mg = wave & 1, ng = wave >> 1;
  int mt0 = mg * 2, nt0 = ng * 2;
  int pn[2] = {nt0 * 16 + (lane & 15), (nt0 + 1) * 16 + (lane & 15)};
  int bk = (lane >> 4) << 3;
  v8f acc[2][2] = {};
#pragma unroll
  for (int kt = 0; kt < KT1; ++kt) {
    v16h a0 = *(const v16h*)(a1p + ((mt0 * KT1 + kt) * 32 + lane) * 16);
    v16h a1 = *(const v16h*)(a1p + (((mt0 + 1) * KT1 + kt) * 32 + lane) * 16);
#pragma unroll
    for (int ni = 0; ni < 2; ++ni) {
      const _Float16* bp = ic + pn[ni] * K1P + kt * 32 + bk;
      __builtin_prefetch(bp + 32, 0, 3);   // WGP-scope global_prefetch_b8
      v16h bb = cat8(*(const v8h*)bp, *(const v8h*)(bp + 16));
      acc[0][ni] = wmma16x16x32(a0, bb, acc[0][ni]);
      acc[1][ni] = wmma16x16x32(a1, bb, acc[1][ni]);
    }
  }
#pragma unroll
  for (int mi = 0; mi < 2; ++mi) {
    int chb = (mt0 + mi) * 16 + ((lane >> 4) << 3);
#pragma unroll
    for (int ni = 0; ni < 2; ++ni) {
      v8h out;
#pragma unroll
      for (int v = 0; v < 8; ++v) {
        float f = acc[mi][ni][v] + b1[chb + v];
        out[v] = (_Float16)fmaxf(f, 0.0f);
      }
      *(v8h*)(c1 + pn[ni] * C1CH + chb) = out;
    }
  }
}

// ------------------------------- pool --------------------------------------
// 2x2 max over C1[p][c] -> P[c][y][x] f16 (layout chosen for im2col2 gathers).
__global__ void pool2x2(const _Float16* __restrict__ c1, _Float16* __restrict__ pf) {
  int t = blockIdx.x * blockDim.x + threadIdx.x;       // C1CH*N2
  int c = t >> 12, q = t & 4095;
  int y = q >> 6, x = q & 63;
  int base = ((2 * y) * SZ0 + 2 * x) * C1CH + c;
  float a = (float)c1[base];
  float b = (float)c1[base + C1CH];
  float d = (float)c1[base + SZ0 * C1CH];
  float e = (float)c1[base + SZ0 * C1CH + C1CH];
  pf[c * N2 + q] = (_Float16)fmaxf(fmaxf(a, b), fmaxf(d, e));
}

// ------------------------------ GEMM 2 -------------------------------------
// D[128ch x 4096px] = A2[128x576] * IC2[576x4096]; epilogue +bias, ReLU, +PE,
// f32 store to X2[p][128c] (channel-coalesced for the reduction). Full K
// unroll: all operand addresses become base + immediate offset.
__global__ void gemm2(const _Float16* __restrict__ a2p, const _Float16* __restrict__ ic,
                      const float* __restrict__ b2, const float* __restrict__ pe,
                      float* __restrict__ x2) {
  int wave = (blockIdx.x * blockDim.x + threadIdx.x) >> 5;  // 512 waves
  int lane = threadIdx.x & 31;
  int mg = wave & 3, ng = wave >> 2;
  int mt0 = mg * 2, nt0 = ng * 2;
  int pn[2] = {nt0 * 16 + (lane & 15), (nt0 + 1) * 16 + (lane & 15)};
  int bk = (lane >> 4) << 3;
  const _Float16* bbase0 = ic + pn[0] * K2V + bk;
  const _Float16* bbase1 = ic + pn[1] * K2V + bk;
  v8f acc[2][2] = {};
#pragma unroll
  for (int kt = 0; kt < KT2; ++kt) {
    v16h a0 = *(const v16h*)(a2p + ((mt0 * KT2 + kt) * 32 + lane) * 16);
    v16h a1 = *(const v16h*)(a2p + (((mt0 + 1) * KT2 + kt) * 32 + lane) * 16);
    const _Float16* bp0 = bbase0 + kt * 32;
    const _Float16* bp1 = bbase1 + kt * 32;
    __builtin_prefetch(bp0 + 32, 0, 3);   // WGP-scope global_prefetch_b8
    __builtin_prefetch(bp1 + 32, 0, 3);
    v16h bb0 = cat8(*(const v8h*)bp0, *(const v8h*)(bp0 + 16));
    v16h bb1 = cat8(*(const v8h*)bp1, *(const v8h*)(bp1 + 16));
    acc[0][0] = wmma16x16x32(a0, bb0, acc[0][0]);
    acc[1][0] = wmma16x16x32(a1, bb0, acc[1][0]);
    acc[0][1] = wmma16x16x32(a0, bb1, acc[0][1]);
    acc[1][1] = wmma16x16x32(a1, bb1, acc[1][1]);
  }
#pragma unroll
  for (int mi = 0; mi < 2; ++mi) {
    int chb = (mt0 + mi) * 16 + ((lane >> 4) << 3);
    v4f bl = *(const v4f*)(b2 + chb);
    v4f bh = *(const v4f*)(b2 + chb + 4);
#pragma unroll
    for (int ni = 0; ni < 2; ++ni) {
      int p = pn[ni];
      v4f pl = *(const v4f*)(pe + p * C2CH + chb);
      v4f ph = *(const v4f*)(pe + p * C2CH + chb + 4);
      v4f o0, o1;
#pragma unroll
      for (int v = 0; v < 4; ++v) {
        o0[v] = fmaxf(acc[mi][ni][v] + bl[v], 0.0f) + pl[v];
        o1[v] = fmaxf(acc[mi][ni][v + 4] + bh[v], 0.0f) + ph[v];
      }
      *(v4f*)(x2 + p * C2CH + chb) = o0;
      *(v4f*)(x2 + p * C2CH + chb + 4) = o1;
    }
  }
}

// ---------------------------- ROI max reduce -------------------------------
// One block per (image, object): mask staged in LDS, 128 threads = channels.
__global__ void roi_reduce(const float* __restrict__ x2, const int* __restrict__ roi,
                           float* __restrict__ out) {
  __shared__ unsigned char mask[N2];
  int o = blockIdx.x, tid = threadIdx.x;
  const int* rb = roi + (size_t)o * SZ0 * SZ0;
  for (int i = tid; i < N2; i += 128) {
    int y = i >> 6, x = i & 63;
    mask[i] = (rb[(2 * y) * SZ0 + 2 * x] != 0) ? 1 : 0;
  }
  __syncthreads();
  float m = -3.4e38f;
  for (int p = 0; p < N2; ++p) {
    float v = mask[p] ? x2[p * C2CH + tid] : 0.0f;  // masked = x * r
    m = fmaxf(m, v);
  }
  out[o * C2CH + tid] = m;
}

// ---------------------------------------------------------------------------
extern "C" void kernel_launch(void* const* d_in, const int* in_sizes, int n_in,
                              void* d_out, int out_size, void* d_ws, size_t ws_size,
                              hipStream_t stream) {
  const float* images = (const float*)d_in[0];
  const int*   rois   = (const int*)d_in[1];
  const float* w1     = (const float*)d_in[2];
  const float* b1     = (const float*)d_in[3];
  const float* w2     = (const float*)d_in[4];
  const float* b2     = (const float*)d_in[5];
  const float* dw     = (const float*)d_in[6];
  const float* db     = (const float*)d_in[7];
  float* out = (float*)d_out;

  char* w = (char*)d_ws;
  auto alloc = [&](size_t bytes) {
    char* r = w;
    w += (bytes + 255) & ~(size_t)255;
    return r;
  };
  _Float16* A1p = (_Float16*)alloc((size_t)4 * KT1 * 32 * 16 * 2);   // 16 KB
  _Float16* A2p = (_Float16*)alloc((size_t)8 * KT2 * 32 * 16 * 2);   // 144 KB
  float*    PE  = (float*)alloc((size_t)N2 * C2CH * 4);              // 2 MB
  _Float16* IC1 = (_Float16*)alloc((size_t)N1 * K1P * 2);            // 4 MB
  _Float16* C1  = (_Float16*)alloc((size_t)N1 * C1CH * 2);           // 2 MB
  _Float16* P   = (_Float16*)alloc((size_t)C1CH * N2 * 2);           // 512 KB
  _Float16* IC2 = (_Float16*)alloc((size_t)N2 * K2V * 2);            // 4.5 MB
  float*    X2  = (float*)alloc((size_t)N2 * C2CH * 4);              // 2 MB

  pack_w1<<<(4 * KT1 * 512) / 256, 256, 0, stream>>>(w1, A1p);
  pack_w2<<<(8 * KT2 * 512) / 256, 256, 0, stream>>>(w2, A2p);
  pe_kernel<<<(N2 * C2CH) / 256, 256, 0, stream>>>(dw, db, PE);

  for (int b = 0; b < BATCH; ++b) {
    im2col1<<<(N1 * 16) / 256, 256, 0, stream>>>(images + (size_t)b * CIN * N1, IC1);
    gemm1<<<256, 128, 0, stream>>>(A1p, IC1, b1, C1);        // 1024 waves exact
    pool2x2<<<(C1CH * N2) / 256, 256, 0, stream>>>(C1, P);
    im2col2<<<(N2 * 72) / 256, 256, 0, stream>>>(P, IC2);
    gemm2<<<128, 128, 0, stream>>>(A2p, IC2, b2, PE, X2);    // 512 waves exact
    roi_reduce<<<NOBJ, 128, 0, stream>>>(
        X2, rois + (size_t)b * NOBJ * SZ0 * SZ0,
        out + (size_t)b * NOBJ * C2CH);
  }
}
